// LGRU_37495064494409
// MI455X (gfx1250) — matrix operand order repaired and checked by reading, not compiled, not run on hardware
//
#include <hip/hip_runtime.h>

// ---------------------------------------------------------------------------
// LGRU persistent-scan kernel for gfx1250 (MI455X).
// bf16 WMMA + fp32 accum; TDM (tensor_load_to_lds) stages h_{t-1} into LDS.
// ---------------------------------------------------------------------------

typedef __attribute__((ext_vector_type(16))) __bf16 v16bf;
typedef __attribute__((ext_vector_type(8)))  __bf16 v8bf;
typedef __attribute__((ext_vector_type(4)))  __bf16 v4bf;
typedef __attribute__((ext_vector_type(8)))  float  v8f;
typedef __attribute__((ext_vector_type(4)))  unsigned int u32x4;
typedef __attribute__((ext_vector_type(8)))  int i32x8;
typedef __attribute__((ext_vector_type(4)))  int i32x4;

constexpr int T_STEPS = 2048;
constexpr int B_ROWS  = 32;
constexpr int F_IN    = 256;
constexpr int H_DIM   = 512;
constexpr int K_DIM   = F_IN + H_DIM;     // 768 fused contraction dim
constexpr int NWG     = 16;               // persistent workgroups
constexpr int COLS    = H_DIM / NWG;      // 32 output columns per WG
constexpr int KT_X    = F_IN / 32;        // 8 k-tiles over x part
constexpr int KT_H    = H_DIM / 32;       // 16 k-tiles over h part

// LDS layout (bf16 elements). Strides padded for bank skew; the Ah pad (8
// bf16 = 16B = 4 DWORDs per 1024B row) is expressible as TDM descriptor
// padding (pad_interval code 7 = 256 DWORDs, pad_amount code 3 = 4 DWORDs).
constexpr int AX_STRIDE = F_IN + 8;       // 264 -> 528B rows
constexpr int AH_STRIDE = H_DIM + 8;      // 520 -> 1040B rows
constexpr int W_STRIDE  = K_DIM + 16;     // 784 -> 1568B cols
constexpr int AX_ELEMS  = B_ROWS * AX_STRIDE;        // 8448
constexpr int AH_ELEMS  = B_ROWS * AH_STRIDE;        // 16640
constexpr int W_ELEMS   = 2 * COLS * W_STRIDE;       // 50176
constexpr unsigned AH_LDS_BYTE_OFF = AX_ELEMS * 2;   // 16896 (dyn LDS base = 0)
constexpr size_t LDS_BYTES = (size_t)(AX_ELEMS + AH_ELEMS + W_ELEMS) * 2; // ~147KB

// workspace layout (bytes)
constexpr size_t WT_ELEMS   = 2ull * H_DIM * K_DIM;  // packed [g][n][k] bf16
constexpr size_t HBUF_ELEMS = 2ull * B_ROWS * H_DIM; // double-buffered h, bf16
constexpr size_t WT_BYTES   = WT_ELEMS * 2;          // 1,572,864
constexpr size_t HBUF_BYTES = HBUF_ELEMS * 2;        // 65,536

// ---------------------------------------------------------------------------
// Pack: Wt[g][n][k] = bf16( k<F ? W_g[k][n] : U_g[k-F][n] ), zero h buffers,
// reset the global barrier counter.
// ---------------------------------------------------------------------------
__global__ void lgru_pack_kernel(const float* __restrict__ Wz,
                                 const float* __restrict__ Wh,
                                 const float* __restrict__ Uz,
                                 const float* __restrict__ Uh,
                                 __bf16* __restrict__ Wt,
                                 __bf16* __restrict__ hbuf,
                                 unsigned* __restrict__ ctr) {
  const int idx = blockIdx.x * blockDim.x + threadIdx.x;
  if (idx < (int)WT_ELEMS) {
    const int g = idx / (H_DIM * K_DIM);
    const int r = idx - g * (H_DIM * K_DIM);
    const int n = r / K_DIM;
    const int k = r - n * K_DIM;
    const float* W = (g == 0) ? Wz : Wh;
    const float* U = (g == 0) ? Uz : Uh;
    const float v = (k < F_IN) ? W[k * H_DIM + n] : U[(k - F_IN) * H_DIM + n];
    Wt[idx] = (__bf16)v;
  }
  if (idx < (int)HBUF_ELEMS) hbuf[idx] = (__bf16)0.0f;  // h0 = 0 (both buffers)
  if (idx == 0) *ctr = 0u;
}

// ---------------------------------------------------------------------------
// TDM descriptor helpers: 2D tile (dim0 x dim1 elements of 2B) -> LDS with
// per-row padding.  Groups per CDNA5 ISA.
// ---------------------------------------------------------------------------
__device__ __forceinline__ u32x4 tdm_group0(unsigned lds_addr,
                                            unsigned long long gaddr) {
  u32x4 g0;
  g0[0] = 1u;                                        // count=1, no gather
  g0[1] = lds_addr;                                  // lds_addr[31:0]
  g0[2] = (unsigned)(gaddr & 0xffffffffull);         // global_addr[31:0]
  g0[3] = (unsigned)((gaddr >> 32) & 0x01ffffffull)  // global_addr[56:32]
          | (2u << 30);                              // type=2 ("image")
  return g0;
}

__device__ __forceinline__ i32x8 tdm_group1_2d(unsigned dim0, unsigned dim1,
                                               unsigned long long stride0,
                                               unsigned tile0, unsigned tile1,
                                               unsigned pad_int_code,
                                               unsigned pad_amt_code) {
  i32x8 g1;
  g1[0] = (int)((1u << 16)                 // data_size = 2B
                | (1u << 20)               // pad_enable
                | (pad_int_code << 22) | (pad_amt_code << 25));
  g1[1] = (int)((dim0 & 0xffffu) << 16);                         // dim0 lo16
  g1[2] = (int)(((dim0 >> 16) & 0xffffu) | ((dim1 & 0xffffu) << 16));
  g1[3] = (int)(((dim1 >> 16) & 0xffffu) | ((tile0 & 0xffffu) << 16));
  g1[4] = (int)(tile1 & 0xffffu);                                // tile_dim2=0
  g1[5] = (int)(unsigned)(stride0 & 0xffffffffull);
  g1[6] = (int)(unsigned)((stride0 >> 32) & 0xffffull);          // stride1=0
  g1[7] = 0;
  return g1;
}

__device__ __forceinline__ void tdm_load_h_tile(unsigned long long gaddr) {
  const u32x4 g0 = tdm_group0(AH_LDS_BYTE_OFF, gaddr);
  const i32x8 g1 = tdm_group1_2d(/*dim0=*/H_DIM, /*dim1=*/B_ROWS,
                                 /*stride0=*/H_DIM, /*tile0=*/H_DIM,
                                 /*tile1=*/B_ROWS,
                                 /*pad_int 1024B=*/7u, /*pad 16B=*/3u);
  const i32x4 z4 = {0, 0, 0, 0};
#if __has_builtin(__builtin_amdgcn_tensor_load_to_lds)
#if defined(__clang_major__) && __clang_major__ >= 23
  const i32x8 z8 = {0, 0, 0, 0, 0, 0, 0, 0};
  __builtin_amdgcn_tensor_load_to_lds(g0, g1, z4, z4, z8, 0);
#else
  __builtin_amdgcn_tensor_load_to_lds(g0, g1, z4, z4, 0);
#endif
#else
  (void)g0; (void)g1; (void)z4;
#endif
}

// ---------------------------------------------------------------------------
// Persistent scan kernel. 16 WGs x 128 threads (4 wave32 waves).
// Wave wv -> C tile (mtile = wv>>1, ntile = wv&1) of this WG's 32 columns,
// computing BOTH gates over the fused K = [x_t | h_{t-1}] contraction.
// ---------------------------------------------------------------------------
__global__ __launch_bounds__(128) void lgru_scan_kernel(
    const float*  __restrict__ x,      // [T,B,F] fp32
    const __bf16* __restrict__ Wt,     // [2][H][K] packed bf16 (col-major K)
    __bf16*       __restrict__ hbuf,   // [2][B][H] bf16 ping-pong
    float*        __restrict__ out,    // [T,B,H] fp32
    unsigned*     __restrict__ ctr) {
  extern __shared__ char smem_raw[];
  __bf16* Axlds = (__bf16*)smem_raw;                // [B_ROWS][AX_STRIDE]
  __bf16* Ahlds = Axlds + AX_ELEMS;                 // [B_ROWS][AH_STRIDE]
  __bf16* Wlds  = Ahlds + AH_ELEMS;                 // [2][COLS][W_STRIDE]

  const int tid  = threadIdx.x;
  const int lane = tid & 31;
  const int wv   = tid >> 5;
  const int mt   = wv >> 1;
  const int nt   = wv & 1;
  const int lrow = lane & 15;
  const bool hiHalf = (lane & 16) != 0;
  const int col0 = blockIdx.x * COLS;

  // ---- Stage this WG's weight slice into LDS once (96KB, hot forever).
  for (int i = tid; i < 64 * 96; i += 128) {
    const int gc = i / 96;
    const int ch = i - gc * 96;
    const int g  = gc >> 5;
    const int c  = gc & 31;
    const uint4 v = *(const uint4*)(Wt + (size_t)g * H_DIM * K_DIM
                                       + (size_t)(col0 + c) * K_DIM + ch * 8);
    *(uint4*)(&Wlds[(g * COLS + c) * W_STRIDE + ch * 8]) = v;
  }
  __syncthreads();

  for (int t = 0; t < T_STEPS; ++t) {
    // ---- TDM: DMA h_{t-1} (32x512 bf16) into padded Ahlds (wave 0 issues).
    const __bf16* hsrc = hbuf + (size_t)(t & 1) * B_ROWS * H_DIM;
    if (tid == 0) {
      tdm_load_h_tile((unsigned long long)(const void*)hsrc);
    }
    // ---- Convert-stage x_t (fp32 -> bf16) into Axlds, all threads.
    const float* xt = x + (size_t)t * B_ROWS * F_IN;
    for (int i = tid; i < B_ROWS * (F_IN / 4); i += 128) {   // 2048 float4s
      const int row = i >> 6;
      const int cc  = i & 63;
      const float4 v = *(const float4*)(xt + row * F_IN + cc * 4);
      v4bf p;
      p[0] = (__bf16)v.x; p[1] = (__bf16)v.y;
      p[2] = (__bf16)v.z; p[3] = (__bf16)v.w;
      *(v4bf*)(&Axlds[row * AX_STRIDE + cc * 4]) = p;
    }
#if !__has_builtin(__builtin_amdgcn_tensor_load_to_lds)
    for (int i = tid; i < B_ROWS * (H_DIM / 8); i += 128) {  // fallback copy
      const int row = i >> 6;
      const int cc  = i & 63;
      const uint4 v = *(const uint4*)(hsrc + row * H_DIM + cc * 8);
      *(uint4*)(&Ahlds[row * AH_STRIDE + cc * 8]) = v;
    }
#endif
    if (t + 1 < T_STEPS)
      __builtin_prefetch((const char*)(x + (size_t)(t + 1) * B_ROWS * F_IN)
                             + tid * 256, 0, 0);
    if (tid == 0) __builtin_amdgcn_s_wait_tensorcnt((short)0);
    __syncthreads();

    // ---- GEMM: each wave accumulates one 16x16 tile for both gates.
    v8f accz = {0.f, 0.f, 0.f, 0.f, 0.f, 0.f, 0.f, 0.f};
    v8f acch = {0.f, 0.f, 0.f, 0.f, 0.f, 0.f, 0.f, 0.f};
    const int arow = mt * 16 + lrow;
    const int bcol = nt * 16 + lrow;
    const __bf16* axp = &Axlds[arow * AX_STRIDE + (hiHalf ? 8 : 0)];
    const __bf16* ahp = &Ahlds[arow * AH_STRIDE + (hiHalf ? 8 : 0)];
    const __bf16* bzb = &Wlds[(0 * COLS + bcol) * W_STRIDE + (hiHalf ? 8 : 0)];
    const __bf16* bhb = &Wlds[(1 * COLS + bcol) * W_STRIDE + (hiHalf ? 8 : 0)];
#pragma unroll 4
    for (int kt = 0; kt < KT_X; ++kt) {               // x-phase: K = 0..255
      const v8bf a0 = *(const v8bf*)(axp + kt * 32);
      const v8bf a1 = *(const v8bf*)(axp + kt * 32 + 16);
      const v16bf a = __builtin_shufflevector(a0, a1, 0, 1, 2, 3, 4, 5, 6, 7,
                                              8, 9, 10, 11, 12, 13, 14, 15);
      const v8bf bz0 = *(const v8bf*)(bzb + kt * 32);
      const v8bf bz1 = *(const v8bf*)(bzb + kt * 32 + 16);
      const v8bf bh0 = *(const v8bf*)(bhb + kt * 32);
      const v8bf bh1 = *(const v8bf*)(bhb + kt * 32 + 16);
      const v16bf bz = __builtin_shufflevector(bz0, bz1, 0, 1, 2, 3, 4, 5, 6, 7,
                                               8, 9, 10, 11, 12, 13, 14, 15);
      const v16bf bh = __builtin_shufflevector(bh0, bh1, 0, 1, 2, 3, 4, 5, 6, 7,
                                               8, 9, 10, 11, 12, 13, 14, 15);
      accz = __builtin_amdgcn_wmma_f32_16x16x32_bf16(
          false, a, false, bz, (short)0, accz, false, false);
      acch = __builtin_amdgcn_wmma_f32_16x16x32_bf16(
          false, a, false, bh, (short)0, acch, false, false);
    }
#pragma unroll 4
    for (int kt = 0; kt < KT_H; ++kt) {               // h-phase: K = 256..767
      const v8bf a0 = *(const v8bf*)(ahp + kt * 32);
      const v8bf a1 = *(const v8bf*)(ahp + kt * 32 + 16);
      const v16bf a = __builtin_shufflevector(a0, a1, 0, 1, 2, 3, 4, 5, 6, 7,
                                              8, 9, 10, 11, 12, 13, 14, 15);
      const v8bf bz0 = *(const v8bf*)(bzb + F_IN + kt * 32);
      const v8bf bz1 = *(const v8bf*)(bzb + F_IN + kt * 32 + 16);
      const v8bf bh0 = *(const v8bf*)(bhb + F_IN + kt * 32);
      const v8bf bh1 = *(const v8bf*)(bhb + F_IN + kt * 32 + 16);
      const v16bf bz = __builtin_shufflevector(bz0, bz1, 0, 1, 2, 3, 4, 5, 6, 7,
                                               8, 9, 10, 11, 12, 13, 14, 15);
      const v16bf bh = __builtin_shufflevector(bh0, bh1, 0, 1, 2, 3, 4, 5, 6, 7,
                                               8, 9, 10, 11, 12, 13, 14, 15);
      accz = __builtin_amdgcn_wmma_f32_16x16x32_bf16(
          false, a, false, bz, (short)0, accz, false, false);
      acch = __builtin_amdgcn_wmma_f32_16x16x32_bf16(
          false, a, false, bh, (short)0, acch, false, false);
    }

    // ---- Epilogue: z = sigmoid(az), hc = relu(ah), h = z*h_prev + (1-z)*hc
    const int rbase = mt * 16 + (hiHalf ? 8 : 0);
    const int gcol  = col0 + nt * 16 + lrow;
    __bf16* hdst = hbuf + (size_t)((t + 1) & 1) * B_ROWS * H_DIM;
    float* outt = out + (size_t)t * B_ROWS * H_DIM;
#pragma unroll
    for (int r = 0; r < 8; ++r) {
      const int b = rbase + r;
      const float az = accz[r];
      const float ah = acch[r];
      const float hp = (float)Ahlds[b * AH_STRIDE + gcol];
      const float z  = 1.0f / (1.0f + __expf(-az));
      const float hc = ah > 0.0f ? ah : 0.0f;
      const float hn = z * hp + (1.0f - z) * hc;
      outt[b * H_DIM + gcol] = hn;
      hdst[b * H_DIM + gcol] = (__bf16)hn;
    }

    // ---- Cross-WG sync: make h_t visible, then monotonic-counter barrier.
    if (t + 1 < T_STEPS) {
      __threadfence();
      __syncthreads();
      __builtin_amdgcn_s_cluster_barrier();   // NOP outside a cluster dispatch
      if (tid == 0) {
        __hip_atomic_fetch_add(ctr, 1u, __ATOMIC_RELEASE,
                               __HIP_MEMORY_SCOPE_AGENT);
        const unsigned target = (unsigned)(NWG * (t + 1));
        while (__hip_atomic_load(ctr, __ATOMIC_ACQUIRE,
                                 __HIP_MEMORY_SCOPE_AGENT) < target) {
          __builtin_amdgcn_s_sleep(2);
        }
      }
      __syncthreads();
      __threadfence();
    }
  }
}

// ---------------------------------------------------------------------------
extern "C" void kernel_launch(void* const* d_in, const int* in_sizes, int n_in,
                              void* d_out, int out_size, void* d_ws,
                              size_t ws_size, hipStream_t stream) {
  const float* x  = (const float*)d_in[0];
  const float* Wz = (const float*)d_in[1];
  const float* Wh = (const float*)d_in[2];
  const float* Uz = (const float*)d_in[3];
  const float* Uh = (const float*)d_in[4];
  float* out = (float*)d_out;

  char* ws = (char*)d_ws;
  __bf16*   Wt   = (__bf16*)ws;                        // 1.5 MB packed weights
  __bf16*   hbuf = (__bf16*)(ws + WT_BYTES);           // 64 KB h ping-pong
  unsigned* ctr  = (unsigned*)(ws + WT_BYTES + HBUF_BYTES);

  const int packThreads = 256;
  const int packBlocks  = (int)((WT_ELEMS + packThreads - 1) / packThreads);
  lgru_pack_kernel<<<packBlocks, packThreads, 0, stream>>>(Wz, Wh, Uz, Uh, Wt,
                                                           hbuf, ctr);
  lgru_scan_kernel<<<NWG, 128, LDS_BYTES, stream>>>(x, Wt, hbuf, out, ctr);
}
